// CAMGATLayer_84782654423352
// MI455X (gfx1250) — matrix operand order repaired and checked by reading, not compiled, not run on hardware
//
#include <hip/hip_runtime.h>
#include <math.h>

#define N_J     21
#define NBATCH  16384
#define NWG     2048
#define PERWG   (NBATCH / NWG)   // 8 batches per workgroup
#define THREADS 256

typedef __attribute__((ext_vector_type(16))) __bf16 v16bf;
typedef __attribute__((ext_vector_type(8)))  __bf16 bf8;
typedef __attribute__((ext_vector_type(8)))  float  v8f;

// f32 -> bf16: let the backend select the native cvt (RNE) instead of a
// hand-rolled 4-op bit sequence.
__device__ __forceinline__ __bf16 f2bf(float f) {
  return (__bf16)f;
}
__device__ __forceinline__ unsigned pack2(float a, float b) {
  unsigned short lo = __builtin_bit_cast(unsigned short, f2bf(a));
  unsigned short hi = __builtin_bit_cast(unsigned short, f2bf(b));
  return (unsigned)lo | ((unsigned)hi << 16);
}

union AFrag { v16bf v; bf8 h[2]; };

// 16-bit A-matrix 16x32 fragment (row-major source, row stride lda elems).
// lanes 0-15: M=lane, K-offsets {0..15}; lanes 16-31: same M, {16..31}.
__device__ __forceinline__ v16bf load_a_frag(const __bf16* tileBase, int lda, int lane) {
  const int m  = lane & 15;
  const int kh = (lane >> 4) << 4;       // 0 or 16
  const __bf16* p = tileBase + m * lda + kh;
  AFrag f;
  f.h[0] = *(const bf8*)(p);             // K kh+0..7
  f.h[1] = *(const bf8*)(p + 8);         // K kh+8..15
  return f.v;
}

// B fragment from pre-swizzled LDS: 16 contiguous bf16 per lane (ntCount==8)
__device__ __forceinline__ v16bf load_b_frag(const __bf16* sw, int ks, int nt, int lane) {
  return *(const v16bf*)(sw + ((((ks << 3) + nt) * 32 + lane) << 4));
}

__device__ __forceinline__ v8f wmma_bf16(v16bf a, v16bf b, v8f c) {
  return __builtin_amdgcn_wmma_f32_16x16x32_bf16(false, a, false, b, (short)0, c,
                                                 false, false);
}

// LDS layout (bytes):
//   W1sw  @ 0       (32768)   W1 128x128 bf16, B-swizzled (KS=4, NT=8)
//   W2sw  @ 32768   (65536)   W2 256x128 bf16, B-swizzled (KS=8, NT=8)
//   a_s   @ 98304   (1024)    a[256] f32
//   b2_s  @ 99328   (512)     bias f32
//   cam_s @ 99840   (1792)    cam 21x21 f32
//   2 batch slots @ 101632 + sb*26880:
//     catbf +0     (16384)  x_cat 32x256 bf16 row-major (cols 0-127 agg, 128-255 x)
//     hsw   +16384 (8192)   h 32(K) x 128(N) bf16 B-swizzled (KS=1, NT=8)
//     alph  +24576 (2048)   alpha 32x32 bf16 row-major, zero padded
//     s_s   +26624 (128)    t_s +26752 (128)
#define LDS_BYTES 155392

extern "C" __global__ void __launch_bounds__(THREADS, 2)
camgat_kernel(const float* __restrict__ x, const float* __restrict__ cam,
              const float* __restrict__ W1, const float* __restrict__ a,
              const float* __restrict__ W2w, const float* __restrict__ W2b,
              float* __restrict__ out)
{
  extern __shared__ char smem[];
  __bf16* W1sw  = (__bf16*)(smem);
  __bf16* W2sw  = (__bf16*)(smem + 32768);
  float*  a_s   = (float*)(smem + 98304);
  float*  b2_s  = (float*)(smem + 99328);
  float*  cam_s = (float*)(smem + 99840);

  const int tid    = threadIdx.x;
  const int lane   = tid & 31;
  const int lanelo = lane & 15;
  const int lanehi = lane >> 4;
  const int sb     = tid >> 7;           // batch slot 0/1
  const int btid   = tid & 127;          // thread id within slot
  const int wv4    = (tid >> 5) & 3;     // wave within slot
  const int mt     = wv4 & 1;            // m-tile (rows 0-15 / 16-31)
  const int ntb    = (wv4 >> 1) << 2;    // n-tile base (0 or 4)

  char*   slot  = smem + 101632 + sb * 26880;
  __bf16* catbf = (__bf16*)(slot);
  __bf16* hsw   = (__bf16*)(slot + 16384);
  __bf16* alph  = (__bf16*)(slot + 24576);
  float*  s_s   = (float*)(slot + 26624);
  float*  t_s   = (float*)(slot + 26752);

  // ---- stage constants (once per WG, amortized over 8 batches) ----
  for (int i = tid; i < 256; i += THREADS) a_s[i]  = a[i];
  for (int i = tid; i < 128; i += THREADS) b2_s[i] = W2b[i];
  for (int i = tid; i < N_J * N_J; i += THREADS) cam_s[i] = cam[i];
  for (int idx = tid; idx < 128 * 128; idx += THREADS) {
    int k = idx >> 7, n = idx & 127;
    int ks = k >> 5, kk = k & 31;
    int li = (n & 15) | ((kk >> 4) << 4);
    W1sw[((((ks << 3) + (n >> 4)) * 32 + li) << 4) + (kk & 15)] = f2bf(W1[idx]);
  }
  for (int idx = tid; idx < 256 * 128; idx += THREADS) {
    int k = idx >> 7, n = idx & 127;
    int ks = k >> 5, kk = k & 31;
    int li = (n & 15) | ((kk >> 4) << 4);
    W2sw[((((ks << 3) + (n >> 4)) * 32 + li) << 4) + (kk & 15)] = f2bf(W2w[idx]);
  }
  __syncthreads();

  const int wgBase = blockIdx.x * PERWG;

  for (int it = 0; it < PERWG / 2; ++it) {
    const int batch = wgBase + it * 2 + sb;

    // ---- P0: x tile -> catbf cols 128..255 (bf16), zero pads/alpha/s/t ----
    {
      const float4* xv = (const float4*)(x + (size_t)batch * (N_J * 128));
      for (int q = btid; q < (N_J * 128) / 4; q += 128) {
        float4 v = xv[q];
        int row = q >> 5, c4 = (q & 31) << 2;
        uint2 pk; pk.x = pack2(v.x, v.y); pk.y = pack2(v.z, v.w);
        *(uint2*)(catbf + row * 256 + 128 + c4) = pk;
      }
      uint2 z; z.x = 0u; z.y = 0u;
      for (int q = btid; q < 352; q += 128) {          // rows 21..31 zero
        int row = 21 + (q >> 5), c4 = (q & 31) << 2;
        *(uint2*)(catbf + row * 256 + 128 + c4) = z;
      }
      for (int q = btid; q < 256; q += 128) ((uint2*)alph)[q] = z;
      if (btid < 32) { s_s[btid] = 0.f; t_s[btid] = 0.f; }
      if (it + 1 < PERWG / 2 && btid < 84)             // next tile: 84 cachelines
        __builtin_prefetch(x + (size_t)(batch + 2) * (N_J * 128) + btid * 32, 0, 0);
    }
    __syncthreads();

    // ---- P1: h = x @ W1 (WMMA bf16), fold s,t out of accumulators, store hsw ----
    {
      v16bf af[4];
      #pragma unroll
      for (int ks = 0; ks < 4; ++ks)
        af[ks] = load_a_frag(catbf + (mt * 16) * 256 + 128 + ks * 32, 256, lane);
      float sp[8], tp[8];
      #pragma unroll
      for (int r = 0; r < 8; ++r) { sp[r] = 0.f; tp[r] = 0.f; }
      #pragma unroll
      for (int q = 0; q < 4; ++q) {
        int nt = ntb + q;
        v8f acc = {0.f, 0.f, 0.f, 0.f, 0.f, 0.f, 0.f, 0.f};
        #pragma unroll
        for (int ks = 0; ks < 4; ++ks)
          acc = wmma_bf16(af[ks], load_b_frag(W1sw, ks, nt, lane), acc);
        int n = (nt << 4) + lanelo;
        float a1v = a_s[n], a2v = a_s[128 + n];
        bf8 hb;
        #pragma unroll
        for (int r = 0; r < 8; ++r) {
          float hv = acc[r];
          sp[r] += hv * a1v;
          tp[r] += hv * a2v;
          hb[r] = f2bf(hv);
        }
        // B-swizzled store: k = mt*16 + lanehi*8 + r, n-col tile nt, lane' = lanelo|mt<<4
        *(bf8*)(hsw + (((nt << 5) + (lanelo | (mt << 4))) << 4) + (lanehi << 3)) = hb;
      }
      #pragma unroll
      for (int r = 0; r < 8; ++r) {
        float sv = sp[r], tv = tp[r];
        #pragma unroll
        for (int msk = 1; msk < 16; msk <<= 1) {      // reduce within 16-lane half
          sv += __shfl_xor(sv, msk, 32);
          tv += __shfl_xor(tv, msk, 32);
        }
        if (lanelo == 0) {
          int row = (mt << 4) + (lanehi << 3) + r;
          atomicAdd(&s_s[row], sv);                   // ds_add_f32
          atomicAdd(&t_s[row], tv);
        }
      }
    }
    __syncthreads();

    // ---- P2: leaky-relu + stable softmax + cam -> alpha (bf16, padded) ----
    if (btid < N_J) {
      int i = btid;
      float si = s_s[i];
      float ebuf[N_J];
      float mx = -1e30f;
      #pragma unroll
      for (int j = 0; j < N_J; ++j) {
        float ev = si + t_s[j];
        ev = ev > 0.f ? ev : 0.2f * ev;
        ebuf[j] = ev;
        mx = fmaxf(mx, ev);
      }
      float sum = 0.f;
      #pragma unroll
      for (int j = 0; j < N_J; ++j) { ebuf[j] = __expf(ebuf[j] - mx); sum += ebuf[j]; }
      float inv = 1.f / sum;
      #pragma unroll
      for (int j = 0; j < N_J; ++j)
        alph[i * 32 + j] = f2bf(cam_s[i * N_J + j] * ebuf[j] * inv);
    }
    __syncthreads();

    // ---- P3: x_agg = alpha @ h (single K-step WMMA) -> catbf cols 0..127 ----
    {
      v16bf aa = load_a_frag(alph + (mt * 16) * 32, 32, lane);
      #pragma unroll
      for (int q = 0; q < 4; ++q) {
        int nt = ntb + q;
        v8f acc = {0.f, 0.f, 0.f, 0.f, 0.f, 0.f, 0.f, 0.f};
        acc = wmma_bf16(aa, load_b_frag(hsw, 0, nt, lane), acc);
        #pragma unroll
        for (int r = 0; r < 8; ++r)
          catbf[((mt << 4) + (lanehi << 3) + r) * 256 + (nt << 4) + lanelo] = f2bf(acc[r]);
      }
    }
    __syncthreads();

    // ---- P4: out = elu(x_cat @ W2 + b2), store 21 real rows ----
    {
      v16bf ac[8];
      #pragma unroll
      for (int ks = 0; ks < 8; ++ks)
        ac[ks] = load_a_frag(catbf + (mt * 16) * 256 + ks * 32, 256, lane);
      #pragma unroll
      for (int q = 0; q < 4; ++q) {
        int nt = ntb + q;
        v8f acc = {0.f, 0.f, 0.f, 0.f, 0.f, 0.f, 0.f, 0.f};
        #pragma unroll
        for (int ks = 0; ks < 8; ++ks)
          acc = wmma_bf16(ac[ks], load_b_frag(W2sw, ks, nt, lane), acc);
        int col = (nt << 4) + lanelo;
        float bias = b2_s[col];
        #pragma unroll
        for (int r = 0; r < 8; ++r) {
          int row = (mt << 4) + (lanehi << 3) + r;
          if (row < N_J) {
            float v = acc[r] + bias;
            v = v > 0.f ? v : (__expf(v) - 1.f);     // ELU(alpha=1)
            out[((size_t)batch * N_J + row) * 128 + col] = v;
          }
        }
      }
    }
    __syncthreads();  // protect LDS before next iteration's P0
  }
}

extern "C" void kernel_launch(void* const* d_in, const int* in_sizes, int n_in,
                              void* d_out, int out_size, void* d_ws, size_t ws_size,
                              hipStream_t stream) {
  const float* x   = (const float*)d_in[0];
  const float* cam = (const float*)d_in[1];
  const float* W1  = (const float*)d_in[2];
  const float* a   = (const float*)d_in[3];
  const float* W2w = (const float*)d_in[4];
  const float* W2b = (const float*)d_in[5];
  float* out = (float*)d_out;

  (void)in_sizes; (void)n_in; (void)out_size; (void)d_ws; (void)ws_size;

  hipFuncSetAttribute((const void*)camgat_kernel,
                      hipFuncAttributeMaxDynamicSharedMemorySize, (int)LDS_BYTES);
  camgat_kernel<<<NWG, THREADS, LDS_BYTES, stream>>>(x, cam, W1, a, W2w, W2b, out);
}
